// Bottleneck_64183991271999
// MI455X (gfx1250) — matrix operand order repaired and checked by reading, not compiled, not run on hardware
//
#include <hip/hip_runtime.h>

typedef __attribute__((ext_vector_type(16))) _Float16 v16h;
typedef __attribute__((ext_vector_type(8)))  _Float16 v8h;
typedef __attribute__((ext_vector_type(8)))  float    v8f;
typedef __attribute__((ext_vector_type(4)))  float    v4f;

#define HW_  3136
#define NT_  6272
#define C_   256
#define EPS_ 1e-5f

#define FLD_ 104   // padded feat row (halves): 52 dwords -> conflict-free LDS stride
#define HLD_ 72    // padded h row (halves):    36 dwords -> conflict-free LDS stride

#define WMMA_F16(A,B,C) __builtin_amdgcn_wmma_f32_16x16x32_f16(false,(A),false,(B),(short)0,(C),false,false)

__device__ __forceinline__ int reflect56(int v) {
    if (v < 0) v = -v;
    if (v >= 56) v = 110 - v;
    return v;
}

// A-operand (16x32 f16), W row-major [M][ld]:
// lane<16: M=m0+lane, K = k0 + {0..7, 16..23}; lane>=16: M=m0+lane-16, K = k0 + {8..15, 24..31}
// Both 8-half chunks are 16B-aligned -> 2 x b128 loads.
__device__ __forceinline__ v16h load_a16(const _Float16* W, int ld, int m0, int k0, int lane) {
    const _Float16* p = W + (size_t)(m0 + (lane & 15)) * ld + k0 + ((lane >> 4) << 3);
    v8h lo = *(const v8h*)p;
    v8h hi = *(const v8h*)(p + 16);
    v16h a;
#pragma unroll
    for (int i = 0; i < 8; ++i) { a[i] = lo[i]; a[8 + i] = hi[i]; }
    return a;
}

// B-operand (32x16 f16) from an N-MAJOR buffer F[N][ld]:
// lane holds N = n0 + (lane&15), K = k0 + (lane>>4)*8 + {0..7, 16..23} -> 2 x b128.
__device__ __forceinline__ v16h load_b16_nm(const _Float16* F, int ld, int k0, int n0, int lane) {
    const _Float16* p = F + (size_t)(n0 + (lane & 15)) * ld + k0 + ((lane >> 4) << 3);
    v8h lo = *(const v8h*)p;
    v8h hi = *(const v8h*)(p + 16);
    v16h b;
#pragma unroll
    for (int i = 0; i < 8; ++i) { b[i] = lo[i]; b[8 + i] = hi[i]; }
    return b;
}

// ---------------- Kernel 0: weight fp32->f16 conversion + position encoding + bn1 fold ----------------
__global__ void k_prep(const float* w1, const float* w2, const float* w3, const float* wc,
                       const float* cw1, const float* cw2, const float* pw, const float* pb,
                       const float* bn1g, const float* bn1b, const float* bn1rm, const float* bn1rv,
                       _Float16* w1h, _Float16* w2h, _Float16* w3h, _Float16* wch,
                       _Float16* cw1h, _Float16* cw2h, float* pbuf, float* sb1, float* tb1) {
    int i = blockIdx.x * 256 + threadIdx.x;
    if (i < 16384) { w1h[i] = (_Float16)w1[i]; return; } i -= 16384;
    if (i < 16384) { w2h[i] = (_Float16)w2[i]; return; } i -= 16384;
    if (i < 65536) { w3h[i] = (_Float16)w3[i]; return; } i -= 65536;
    if (i < 65536) { wch[i] = (_Float16)wc[i]; return; } i -= 65536;
    if (i < 6144)  { int o = i / 96, k = i % 96;            // pad K 66 -> 96 with zeros
                     cw1h[i] = (_Float16)(k < 66 ? cw1[o * 66 + k] : 0.f); return; } i -= 6144;
    if (i < 2048)  { cw2h[i] = (_Float16)cw2[i]; return; } i -= 2048;
    if (i < 2 * HW_) {
        int j = i / HW_, q = i % HW_;
        int h = q / 56, w = q % 56;
        float lw = -1.f + 2.f * (float)w / 55.f;
        float lh = -1.f + 2.f * (float)h / 55.f;
        pbuf[i] = pw[j * 2 + 0] * lw + pw[j * 2 + 1] * lh + pb[j];
        return;
    } i -= 2 * HW_;
    if (i < 256) {
        float s = bn1g[i] * rsqrtf(bn1rv[i] + EPS_);
        sb1[i] = s; tb1[i] = bn1b[i] - bn1rm[i] * s;
    }
}

// ---------------- Kernel 1: bn1 + relu -> f16 activations, N-MAJOR [N][C] ----------------
__global__ void k_act(const float* x, const float* sb1, const float* tb1, _Float16* act) {
    int idx = blockIdx.x * 256 + threadIdx.x;   // NT_*32 threads, 8 channels each
    int n = idx >> 5, cb = (idx & 31) << 3;
    int b = n / HW_, q = n % HW_;
    v4f s0 = *(const v4f*)(sb1 + cb), s1 = *(const v4f*)(sb1 + cb + 4);
    v4f t0 = *(const v4f*)(tb1 + cb), t1 = *(const v4f*)(tb1 + cb + 4);
    v8h out;
#pragma unroll
    for (int i = 0; i < 8; ++i) {
        float xv = x[((size_t)b * C_ + cb + i) * HW_ + q];
        float s = (i < 4) ? s0[i] : s1[i - 4];
        float t = (i < 4) ? t0[i] : t1[i - 4];
        out[i] = (_Float16)fmaxf(xv * s + t, 0.f);
    }
    *(v8h*)(act + (size_t)n * C_ + cb) = out;
}

// ---------------- Kernel 2: WMMA projections x1/x2 ([64][N] f32), x3 ([N][256] f32) ----------------
__global__ void k_proj(const _Float16* w1h, const _Float16* w2h, const _Float16* w3h,
                       const _Float16* act, const float* b1, const float* b2, const float* b3,
                       float* X1, float* X2, float* X3T) {
    int lane = threadIdx.x & 31;
    int nt = blockIdx.x, mt = blockIdx.y;
    const _Float16* W; int m0, which;
    if (mt < 4)      { W = w1h; m0 = mt * 16;       which = 0; }
    else if (mt < 8) { W = w2h; m0 = (mt - 4) * 16; which = 1; }
    else             { W = w3h; m0 = (mt - 8) * 16; which = 2; }
    __builtin_prefetch(W + (size_t)m0 * 256, 0, 3);
    v8f acc = {};
#pragma unroll
    for (int ks = 0; ks < 8; ++ks) {
        v16h a  = load_a16(W, 256, m0, ks * 32, lane);
        v16h bv = load_b16_nm(act, C_, ks * 32, nt * 16, lane);
        acc = WMMA_F16(a, bv, acc);
    }
    int n = nt * 16 + (lane & 15);
    int mb = m0 + ((lane >> 4) << 3);
#pragma unroll
    for (int r = 0; r < 8; ++r) {
        int m = mb + r;
        if (which == 0)      X1[(size_t)m * NT_ + n] = acc[r] + b1[m];
        else if (which == 1) X2[(size_t)m * NT_ + n] = acc[r] + b2[m];
        else                 X3T[(size_t)n * 256 + m] = acc[r] + b3[m];
    }
}

// ---------------- Kernel 3: fused feat -> cw1 -> cw2 -> softmax -> aggregation ----------------
// One workgroup (8 waves) handles 16 pixels of one batch; everything lives in LDS.
__global__ __launch_bounds__(256) void k_san(
    const float* X1, const float* X2, const float* X3T, const float* pbuf,
    const _Float16* cw1h, const _Float16* cw2h,
    const float* c1g, const float* c1b, const float* c1rm, const float* c1rv,
    const float* c2g, const float* c2b, const float* c2rm, const float* c2rv,
    const float* cb2,
    const float* b2g, const float* b2b, const float* b2rm, const float* b2rv,
    _Float16* aggn) {
    extern __shared__ char smem[];
    _Float16* featS = (_Float16*)smem;                                  // [784][FLD_] f16, pos-major
    float*    wS    = (float*)(smem + 784 * FLD_ * 2);                  // [49][16][32] f32
    _Float16* hS    = (_Float16*)((char*)wS + 49 * 512 * 4);            // 8 waves * [16][HLD_]
    int*      nqS   = (int*)((char*)hS + 8 * 16 * HLD_ * 2);            // [49*16]
    float*    s1S   = (float*)(nqS + 784);
    float*    t1S   = s1S + 66;
    float*    s2S   = t1S + 66;
    float*    t2S   = s2S + 64;
    float*    sbS   = t2S + 64;
    float*    tbS   = sbS + 256;

    int tid = threadIdx.x;
    int b = blockIdx.x / 196;
    int qbase = (blockIdx.x % 196) * 16;

    if (tid < 66) { float s = c1g[tid] * rsqrtf(c1rv[tid] + EPS_); s1S[tid] = s; t1S[tid] = c1b[tid] - c1rm[tid] * s; }
    if (tid < 64) { float s = c2g[tid] * rsqrtf(c2rv[tid] + EPS_); s2S[tid] = s; t2S[tid] = c2b[tid] - c2rm[tid] * s; }
    { float s = b2g[tid] * rsqrtf(b2rv[tid] + EPS_); sbS[tid] = s; tbS[tid] = b2b[tid] - b2rm[tid] * s; }

    for (int t = tid; t < 784; t += 256) {                 // reflect-pad neighbor indices
        int k = t >> 4, pix = t & 15;
        int q = qbase + pix;
        int rr = reflect56(q / 56 + k / 7 - 3);
        int cc = reflect56(q % 56 + k % 7 - 3);
        nqS[t] = rr * 56 + cc;
    }
    __syncthreads();

    // build normalized feat tile (cwbn1 folded + relu); channels 66..FLD_-1 zero-padded.
    // ch-outer mapping keeps the X1/X2/pbuf global reads coalesced across pos.
    for (int e = tid; e < FLD_ * 784; e += 256) {
        int ch = e / 784, pos = e % 784;
        _Float16 outv = (_Float16)0.f;
        if (ch < 66) {
            int pix = pos & 15;
            int q = qbase + pix;
            int nq = nqS[pos];
            float v;
            if (ch < 64) v = X1[(size_t)ch * NT_ + b * HW_ + q] - X2[(size_t)ch * NT_ + b * HW_ + nq];
            else { int j = ch - 64; v = pbuf[j * HW_ + q] - pbuf[j * HW_ + nq]; }
            outv = (_Float16)fmaxf(s1S[ch] * v + t1S[ch], 0.f);
        }
        featS[pos * FLD_ + ch] = outv;
    }
    __syncthreads();

    int wave = tid >> 5, lane = tid & 31;
    _Float16* hw = hS + wave * (16 * HLD_);
    int mb  = (lane >> 4) << 3;
    int col = lane & 15;
    for (int j = wave; j < 49; j += 8) {                   // j == kernel tap k
        v8f acc[4] = {{}, {}, {}, {}};
#pragma unroll
        for (int ks = 0; ks < 3; ++ks) {                   // K = 96 : 66 ch + zero pad
            v16h bv = load_b16_nm(featS, FLD_, ks * 32, j * 16, lane);
#pragma unroll
            for (int t = 0; t < 4; ++t) {
                v16h a = load_a16(cw1h, 96, t * 16, ks * 32, lane);
                acc[t] = WMMA_F16(a, bv, acc[t]);
            }
        }
        // cwbn2 fold + relu; re-swizzle D-layout -> B-layout via per-wave LDS (packed b128 stores)
#pragma unroll
        for (int t = 0; t < 4; ++t) {
            v8h hv;
#pragma unroll
            for (int r = 0; r < 8; ++r) {
                int m = t * 16 + mb + r;
                hv[r] = (_Float16)fmaxf(s2S[m] * acc[t][r] + t2S[m], 0.f);
            }
            *(v8h*)(hw + col * HLD_ + t * 16 + mb) = hv;
        }
        v8f lg[2] = {{}, {}};
#pragma unroll
        for (int ks = 0; ks < 2; ++ks) {                   // K = 64
            v16h bv = load_b16_nm(hw, HLD_, ks * 32, 0, lane);
#pragma unroll
            for (int t = 0; t < 2; ++t) {
                v16h a = load_a16(cw2h, 64, t * 16, ks * 32, lane);
                lg[t] = WMMA_F16(a, bv, lg[t]);
            }
        }
        // logits + bias -> wS[k][pix][hn], contiguous per accumulator -> 2 x v4f stores
#pragma unroll
        for (int t = 0; t < 2; ++t) {
            float* dst = wS + j * 512 + col * 32 + t * 16 + mb;
            v4f lo, hi;
#pragma unroll
            for (int r = 0; r < 4; ++r) { lo[r] = lg[t][r] + cb2[t * 16 + mb + r];
                                          hi[r] = lg[t][r + 4] + cb2[t * 16 + mb + r + 4]; }
            *(v4f*)dst = lo;
            *(v4f*)(dst + 4) = hi;
        }
    }
    __syncthreads();

    // softmax over the 49-tap footprint, per (pix, hn); hn-inner -> conflict-free
    for (int u = tid; u < 512; u += 256) {
        int pix = u >> 5, hn = u & 31;
        float* p = wS + pix * 32 + hn;
        float mx = -1e30f;
        for (int k = 0; k < 49; ++k) mx = fmaxf(mx, p[k * 512]);
        float s = 0.f;
        for (int k = 0; k < 49; ++k) { float e = __expf(p[k * 512] - mx); p[k * 512] = e; s += e; }
        float inv = 1.f / s;
        for (int k = 0; k < 49; ++k) p[k * 512] *= inv;
    }
    __syncthreads();

    // aggregation over unfolded x3 (share=8), bn2 fold + relu -> f16, N-MAJOR [N][C]
    int c = tid, hn = c >> 3;
    for (int pix = 0; pix < 16; ++pix) {
        float accv = 0.f;
        for (int k = 0; k < 49; ++k)
            accv += wS[k * 512 + pix * 32 + hn] * X3T[(size_t)(b * HW_ + nqS[k * 16 + pix]) * 256 + c];
        float v = fmaxf(sbS[c] * accv + tbS[c], 0.f);
        aggn[(size_t)(b * HW_ + qbase + pix) * C_ + c] = (_Float16)v;
    }
}

// ---------------- Kernel 4: WMMA final GEMM wc + bias + identity ----------------
__global__ void k_final(const _Float16* wch, const _Float16* aggn, const float* bc,
                        const float* x, float* y) {
    int lane = threadIdx.x & 31;
    int nt = blockIdx.x, mt = blockIdx.y;
    __builtin_prefetch(wch + (size_t)mt * 16 * 256, 0, 3);
    v8f acc = {};
#pragma unroll
    for (int ks = 0; ks < 8; ++ks) {
        v16h a  = load_a16(wch, 256, mt * 16, ks * 32, lane);
        v16h bv = load_b16_nm(aggn, C_, ks * 32, nt * 16, lane);
        acc = WMMA_F16(a, bv, acc);
    }
    int n = nt * 16 + (lane & 15);
    int b = n / HW_, q = n % HW_;
    int mb = mt * 16 + ((lane >> 4) << 3);
#pragma unroll
    for (int r = 0; r < 8; ++r) {
        int m = mb + r;
        size_t idx = ((size_t)b * C_ + m) * HW_ + q;
        y[idx] = acc[r] + bc[m] + x[idx];
    }
}

extern "C" void kernel_launch(void* const* d_in, const int* in_sizes, int n_in,
                              void* d_out, int out_size, void* d_ws, size_t ws_size,
                              hipStream_t stream) {
    (void)in_sizes; (void)n_in; (void)out_size; (void)ws_size;
    const float* x       = (const float*)d_in[0];
    const float* bn1_g   = (const float*)d_in[1];
    const float* bn1_b   = (const float*)d_in[2];
    const float* bn1_rm  = (const float*)d_in[3];
    const float* bn1_rv  = (const float*)d_in[4];
    const float* w1      = (const float*)d_in[5];
    const float* b1      = (const float*)d_in[6];
    const float* w2      = (const float*)d_in[7];
    const float* b2      = (const float*)d_in[8];
    const float* w3      = (const float*)d_in[9];
    const float* b3      = (const float*)d_in[10];
    const float* pw      = (const float*)d_in[11];
    const float* pb      = (const float*)d_in[12];
    const float* c1g     = (const float*)d_in[13];
    const float* c1b     = (const float*)d_in[14];
    const float* c1rm    = (const float*)d_in[15];
    const float* c1rv    = (const float*)d_in[16];
    const float* cw1     = (const float*)d_in[17];
    const float* c2g     = (const float*)d_in[18];
    const float* c2b     = (const float*)d_in[19];
    const float* c2rm    = (const float*)d_in[20];
    const float* c2rv    = (const float*)d_in[21];
    const float* cw2     = (const float*)d_in[22];
    const float* cb2     = (const float*)d_in[23];
    const float* b2g     = (const float*)d_in[24];
    const float* b2b     = (const float*)d_in[25];
    const float* b2rm    = (const float*)d_in[26];
    const float* b2rv    = (const float*)d_in[27];
    const float* wc      = (const float*)d_in[28];
    const float* bc      = (const float*)d_in[29];

    char* ws = (char*)d_ws;
    size_t off = 0;
    auto alloc = [&](size_t bytes) -> void* {
        void* p = ws + off;
        off = (off + bytes + 255) & ~(size_t)255;
        return p;
    };
    _Float16* acth = (_Float16*)alloc((size_t)NT_ * C_ * 2);   // [N][C]
    _Float16* w1h  = (_Float16*)alloc(64 * 256 * 2);
    _Float16* w2h  = (_Float16*)alloc(64 * 256 * 2);
    _Float16* w3h  = (_Float16*)alloc(256 * 256 * 2);
    _Float16* wch  = (_Float16*)alloc(256 * 256 * 2);
    _Float16* cw1h = (_Float16*)alloc(64 * 96 * 2);
    _Float16* cw2h = (_Float16*)alloc(32 * 64 * 2);
    float*    pbuf = (float*)alloc(2 * HW_ * 4);
    float*    sb1  = (float*)alloc(256 * 4);
    float*    tb1  = (float*)alloc(256 * 4);
    float*    X1   = (float*)alloc((size_t)64 * NT_ * 4);      // [64][N]
    float*    X2   = (float*)alloc((size_t)64 * NT_ * 4);      // [64][N]
    float*    X3T  = (float*)alloc((size_t)NT_ * 256 * 4);     // [N][C]
    _Float16* aggn = (_Float16*)alloc((size_t)NT_ * C_ * 2);   // [N][C]

    k_prep<<<698, 256, 0, stream>>>(w1, w2, w3, wc, cw1, cw2, pw, pb,
                                    bn1_g, bn1_b, bn1_rm, bn1_rv,
                                    w1h, w2h, w3h, wch, cw1h, cw2h, pbuf, sb1, tb1);
    k_act<<<(NT_ * 32) / 256, 256, 0, stream>>>(x, sb1, tb1, acth);
    k_proj<<<dim3(392, 24), 32, 0, stream>>>(w1h, w2h, w3h, acth, b1, b2, b3, X1, X2, X3T);

    size_t smem = 784 * FLD_ * 2 + 49 * 512 * 4 + 8 * 16 * HLD_ * 2 + 784 * 4
                + (66 + 66 + 64 + 64 + 256 + 256) * 4;   // 288,080 B < 320 KB WGP LDS
    k_san<<<392, 256, smem, stream>>>(X1, X2, X3T, pbuf, cw1h, cw2h,
                                      c1g, c1b, c1rm, c1rv,
                                      c2g, c2b, c2rm, c2rv, cb2,
                                      b2g, b2b, b2rm, b2rv, aggn);
    k_final<<<dim3(392, 16), 32, 0, stream>>>(wch, aggn, bc, x, (float*)d_out);
}
